// RNN_22213570855355
// MI455X (gfx1250) — matrix operand order repaired and checked by reading, not compiled, not run on hardware
//
#include <hip/hip_runtime.h>
#include <math.h>

// ---------------------------------------------------------------------------
// Problem constants
#define Bn 1024
#define Sn 256
#define Dn 300
#define Hn 150
#define KP 320      // padded input K (300 -> 320), bf16 row pitch for A matrices
#define NP 640      // padded gate width (600 -> 640)
#define HKP 160     // padded recurrent K (150 -> 160)
#define LASTP 304   // row pitch of 'last' buffer (f32)

typedef __attribute__((ext_vector_type(16))) __bf16 bf16x16;
typedef __attribute__((ext_vector_type(8)))  float  f32x8;

union FragB { unsigned int u[8]; bf16x16 v; };

__device__ __forceinline__ unsigned short f2bf(float f) {
    unsigned int u = __float_as_uint(f);
    u += 0x7FFFu + ((u >> 16) & 1u);   // round-to-nearest-even
    return (unsigned short)(u >> 16);
}

__device__ __forceinline__ void async_b128(unsigned ldsAddr, unsigned gOff,
                                           const void* saddr) {
    // GVS form: mem = SADDR + VGPR_off + IOFFSET ; lds = VGPR_lds + IOFFSET
    asm volatile("global_load_async_to_lds_b128 %0, %1, %2 offset:0"
                 :: "v"(ldsAddr), "v"(gOff), "s"(saddr) : "memory");
    asm volatile("global_load_async_to_lds_b128 %0, %1, %2 offset:16"
                 :: "v"(ldsAddr), "v"(gOff), "s"(saddr) : "memory");
}

__device__ __forceinline__ void wait_async0() {
    asm volatile("s_wait_asynccnt 0x0" ::: "memory");
}

__device__ __forceinline__ unsigned lds_off32(const void* p) {
    // Generic LDS pointer: addr[31:0] is the LDS byte offset (aperture in high bits)
    return (unsigned)(unsigned long long)p;
}

// ---------------------------------------------------------------------------
// lengths[b] = sum(mask[b, :] != 0)
__global__ void k_lengths(const int* __restrict__ mask, int* __restrict__ lengths) {
    int row = blockIdx.x, t = threadIdx.x;
    int pred = mask[row * Sn + t] != 0;
    unsigned long long b = __ballot(pred);
    __shared__ int part[8];
    if ((t & 31) == 0) part[t >> 5] = __popcll(b);
    __syncthreads();
    if (t == 0) {
        int s = 0;
        for (int i = 0; i < 8; i++) s += part[i];
        lengths[row] = s;
    }
}

// ---------------------------------------------------------------------------
// Embedding gather -> bf16, padded to KP, with emb row 0 zeroed.
__global__ void k_embed(const int* __restrict__ x, const float* __restrict__ emb,
                        unsigned short* __restrict__ xe, long total) {
    long e = (long)blockIdx.x * blockDim.x + threadIdx.x;
    if (e >= total) return;
    int  col = (int)(e % KP);
    long row = e / KP;
    int  tok = x[row];
    float v = 0.f;
    if (col < Dn && tok != 0) v = emb[(long)tok * Dn + col];
    xe[e] = f2bf(v);
}

// ---------------------------------------------------------------------------
// f32 [rows, cols] -> bf16 [640, dcols] zero-padded
__global__ void k_padw(const float* __restrict__ src, unsigned short* __restrict__ dst,
                       int rows, int cols, int dcols) {
    int e = blockIdx.x * blockDim.x + threadIdx.x;
    if (e >= 640 * dcols) return;
    int r = e / dcols, c = e - r * dcols;
    float v = (r < rows && c < cols) ? src[r * cols + c] : 0.f;
    dst[e] = f2bf(v);
}

__global__ void k_padb(const float* __restrict__ src, float* __restrict__ dst) {
    int e = blockIdx.x * blockDim.x + threadIdx.x;
    if (e < NP) dst[e] = (e < 600) ? src[e] : 0.f;
}

__global__ void k_zero32(unsigned int* __restrict__ p, long n) {
    long e = (long)blockIdx.x * blockDim.x + threadIdx.x;
    if (e < n) p[e] = 0u;
}

// ---------------------------------------------------------------------------
// C[M, NP](f32) = A[M, KP](bf16) @ W[NP, KP]^T(bf16) + bias[NP]
// 128x128 macro tile, 8 waves, each wave: 16(M) x 128(N), K-step 32.
// Double-buffered LDS tiles staged with async-to-LDS B128; one barrier/K-step.
// B-fragment pipeline depth 2 -> partial dscnt waits under the WMMAs.
// Tile pitch = 40 halfwords (80 B): 16B-aligned rows for B128, conflict-free frags.
#define TP  40          // tile pitch in bf16
#define TPW 20          // tile pitch in words
#define TILE_BYTES (128 * TP * 2)

__global__ void __launch_bounds__(256) k_gemm_in(
    const unsigned short* __restrict__ A,
    const unsigned short* __restrict__ W,
    const float* __restrict__ biasP,
    float* __restrict__ C, int M)
{
    __shared__ unsigned short As[2][128 * TP];
    __shared__ unsigned short Bs[2][128 * TP];
    const int tid   = threadIdx.x;
    const int mBase = blockIdx.x * 128;
    const int nBase = blockIdx.y * 128;
    const int w     = tid >> 5, l = tid & 31;
    const int khalf = l >> 4, lm = l & 15;

    f32x8 acc[8];
    #pragma unroll
    for (int nt = 0; nt < 8; nt++) {
        float bv = biasP[nBase + nt * 16 + lm];
        #pragma unroll
        for (int r = 0; r < 8; r++) acc[nt][r] = bv;
    }

    // Staging: thread -> (row lr, half lh), 32 B each side per K-step.
    const int lr = tid >> 1, lh = tid & 1;
    const unsigned ldsA0 = lds_off32(&As[0][0]) + lr * (TP * 2) + lh * 32;
    const unsigned ldsB0 = lds_off32(&Bs[0][0]) + lr * (TP * 2) + lh * 32;
    const unsigned gOffA0 = (unsigned)(((mBase + lr) * KP + lh * 16) * 2);
    const unsigned gOffW0 = (unsigned)(((nBase + lr) * KP + lh * 16) * 2);

    // Prologue: stage K-step 0 into buffer 0
    async_b128(ldsA0, gOffA0, A);
    async_b128(ldsB0, gOffW0, W);
    wait_async0();
    __syncthreads();

    for (int kb = 0; kb < 10; kb++) {        // K = 320 = 10 * 32
        const int cur = kb & 1;
        if (kb < 9) {                        // stage next K-step into other buffer
            const unsigned bo = (unsigned)((cur ^ 1) * TILE_BYTES);
            const unsigned go = (unsigned)((kb + 1) * 64);
            async_b128(ldsA0 + bo, gOffA0 + go, A);
            async_b128(ldsB0 + bo, gOffW0 + go, W);
        }

        const unsigned int* AW = (const unsigned int*)As[cur];
        const unsigned int* BW = (const unsigned int*)Bs[cur];
        FragB a, b0, b1, b2;
        #pragma unroll
        for (int j = 0; j < 8; j++)    // A 16x32: lane m=l&15, khalf=l>>4
            a.u[j] = AW[(w * 16 + lm) * TPW + j + khalf * 8];
        #pragma unroll
        for (int j = 0; j < 8; j++)    // B 32x16 tile 0
            b0.u[j] = BW[lm * TPW + j + khalf * 8];
        #pragma unroll
        for (int j = 0; j < 8; j++)    // B tile 1
            b1.u[j] = BW[(16 + lm) * TPW + j + khalf * 8];
        #pragma unroll
        for (int nt = 0; nt < 8; nt++) {
            if (nt < 6) {              // prefetch frag nt+2 -> 2 frags in flight
                #pragma unroll
                for (int j = 0; j < 8; j++)
                    b2.u[j] = BW[((nt + 2) * 16 + lm) * TPW + j + khalf * 8];
            }
            acc[nt] = __builtin_amdgcn_wmma_f32_16x16x32_bf16(
                false, a.v, false, b0.v, (short)0, acc[nt], false, false);
            b0 = b1;
            b1 = b2;
        }

        wait_async0();                 // next tile landed while we computed
        __syncthreads();
    }

    #pragma unroll
    for (int nt = 0; nt < 8; nt++) {
        int ncol = nBase + nt * 16 + lm;
        #pragma unroll
        for (int r = 0; r < 8; r++) {
            int mrow = mBase + w * 16 + r + khalf * 8;
            C[(long)mrow * NP + ncol] = acc[nt][r];
        }
    }
}

// ---------------------------------------------------------------------------
// Persistent recurrent scan. Block = 16 batch rows, 8 waves.
// Whh (bf16, padded [640][160], LDS pitch 164) resident in LDS (~210 KB).
// mode 0: write masked sequence output to h1 [B*S][KP] (col dir*150+j).
// mode 1: fwd writes final h, bwd writes step-0 output, to last[B][LASTP].
#define SMEM_WHH   0
#define SMEM_HBF   209920
#define SMEM_HF    215168
#define SMEM_CF    224896
#define SMEM_GATES 234624
#define SMEM_LS    275840
#define SMEM_TOTAL 275904

__global__ void __launch_bounds__(256) k_scan(
    const float* __restrict__ Gf, const float* __restrict__ Gb,
    const unsigned short* __restrict__ WhhF, const unsigned short* __restrict__ WhhB,
    const int* __restrict__ lengths,
    unsigned short* __restrict__ h1, float* __restrict__ last, int mode)
{
    extern __shared__ char smem[];
    unsigned int*   whhW = (unsigned int*)(smem + SMEM_WHH);    // [640][82] words
    unsigned int*   hbfW = (unsigned int*)(smem + SMEM_HBF);    // [16][82] words
    unsigned short* hbfS = (unsigned short*)(smem + SMEM_HBF);  // [16][164] bf16
    float* hf    = (float*)(smem + SMEM_HF);                    // [16][152]
    float* cf    = (float*)(smem + SMEM_CF);                    // [16][152]
    float* gates = (float*)(smem + SMEM_GATES);                 // [16][644]
    int*   Ls    = (int*)(smem + SMEM_LS);

    const int tid = threadIdx.x;
    const int dir = blockIdx.y;
    const int b0  = blockIdx.x * 16;
    const float* G = dir ? Gb : Gf;
    const unsigned short* Whh = dir ? WhhB : WhhF;

    // Stage Whh into LDS once (read from HBM once per block).
    const unsigned int* gW = (const unsigned int*)Whh;
    for (int idx = tid; idx < 640 * 80; idx += 256) {
        int n = idx / 80, c = idx - n * 80;
        whhW[n * 82 + c] = gW[idx];
    }
    for (int i = tid; i < 16 * 82; i += 256) hbfW[i] = 0u;
    for (int i = tid; i < 16 * 152; i += 256) { hf[i] = 0.f; cf[i] = 0.f; }
    if (tid < 16) Ls[tid] = lengths[b0 + tid];
    __syncthreads();

    const int w = tid >> 5, l = tid & 31;
    const int khalf = l >> 4, lm = l & 15;
    const int nb = w * 5;                     // wave owns N-tiles nb..nb+4

    for (int t = 0; t < Sn; t++) {
        // ---- recurrent GEMM: gates_rec[16x640] = h[16x160] @ Whh^T ----
        f32x8 acc[5] = {};
        FragB a, an, b, bn;
        #pragma unroll
        for (int j = 0; j < 8; j++)
            a.u[j] = hbfW[lm * 82 + j + khalf * 8];
        #pragma unroll
        for (int j = 0; j < 8; j++)
            b.u[j] = whhW[(nb * 16 + lm) * 82 + j + khalf * 8];
        #pragma unroll
        for (int kk = 0; kk < 5; kk++) {      // K = 160 = 5 * 32
            #pragma unroll
            for (int tt = 0; tt < 5; tt++) {
                if (tt < 4) {                 // prefetch next B frag (same kk)
                    #pragma unroll
                    for (int j = 0; j < 8; j++)
                        bn.u[j] = whhW[((nb + tt + 1) * 16 + lm) * 82 + kk * 16 + j + khalf * 8];
                } else if (kk < 4) {          // prefetch A and B frag of next kk
                    #pragma unroll
                    for (int j = 0; j < 8; j++)
                        an.u[j] = hbfW[lm * 82 + (kk + 1) * 16 + j + khalf * 8];
                    #pragma unroll
                    for (int j = 0; j < 8; j++)
                        bn.u[j] = whhW[(nb * 16 + lm) * 82 + (kk + 1) * 16 + j + khalf * 8];
                }
                acc[tt] = __builtin_amdgcn_wmma_f32_16x16x32_bf16(
                    false, a.v, false, b.v, (short)0, acc[tt], false, false);
                b = bn;
            }
            a = an;
        }
        #pragma unroll
        for (int tt = 0; tt < 5; tt++) {
            int ncol = (nb + tt) * 16 + lm;
            #pragma unroll
            for (int r = 0; r < 8; r++)
                gates[(r + khalf * 8) * 644 + ncol] = acc[tt][r];
        }
        __syncthreads();

        // ---- elementwise LSTM cell with packed-sequence masking ----
        for (int e = tid; e < 16 * Hn; e += 256) {
            int m = e / Hn, j = e - m * Hn;
            int L = Ls[m];
            bool valid = t < L;
            int tm = dir ? (valid ? (L - 1 - t) : t) : t;
            long grow = ((long)(b0 + m) * Sn + tm) * NP;
            float gi = gates[m * 644 + j]         + G[grow + j];
            float gfv= gates[m * 644 + 150 + j]   + G[grow + 150 + j];
            float gg = gates[m * 644 + 300 + j]   + G[grow + 300 + j];
            float go = gates[m * 644 + 450 + j]   + G[grow + 450 + j];
            float i_ = 1.f / (1.f + __expf(-gi));
            float f_ = 1.f / (1.f + __expf(-gfv));
            float g_ = tanhf(gg);
            float o_ = 1.f / (1.f + __expf(-go));
            if (valid) {
                float cn = f_ * cf[m * 152 + j] + i_ * g_;
                float hn = o_ * tanhf(cn);
                cf[m * 152 + j] = cn;
                hf[m * 152 + j] = hn;
                hbfS[m * 164 + j] = f2bf(hn);
                if (mode == 0) {
                    h1[(((long)(b0 + m) * Sn + tm) * KP) + dir * Hn + j] = f2bf(hn);
                } else if (dir == 1 && t == 0) {
                    last[(b0 + m) * LASTP + Hn + j] = hn;  // bwd output at pos L-1
                }
            }
        }
        __syncthreads();
    }
    if (mode == 1 && dir == 0) {               // fwd final (frozen) state
        for (int e = tid; e < 16 * Hn; e += 256) {
            int m = e / Hn, j = e - m * Hn;
            last[(b0 + m) * LASTP + j] = hf[m * 152 + j];
        }
    }
}

// ---------------------------------------------------------------------------
// out = tanh(last @ W1^T + b1) @ W2^T + b2 ; one block per batch row.
__global__ void __launch_bounds__(320) k_head(
    const float* __restrict__ last, const float* __restrict__ W1, const float* __restrict__ b1,
    const float* __restrict__ W2, const float* __restrict__ b2, float* __restrict__ out)
{
    __shared__ float xr[304];
    __shared__ float t1[304];
    int row = blockIdx.x, t = threadIdx.x;
    if (t < Dn) xr[t] = last[row * LASTP + t];
    __syncthreads();
    if (t < Dn) {
        float s = b1[t];
        const float* wr = W1 + t * Dn;
        for (int k = 0; k < Dn; k++) s += xr[k] * wr[k];
        t1[t] = tanhf(s);
    }
    __syncthreads();
    if (t < Dn) {
        float s = b2[t];
        const float* wr = W2 + t * Dn;
        for (int k = 0; k < Dn; k++) s += t1[k] * wr[k];
        out[row * Dn + t] = s;
    }
}

// ---------------------------------------------------------------------------
extern "C" void kernel_launch(void* const* d_in, const int* in_sizes, int n_in,
                              void* d_out, int out_size, void* d_ws, size_t ws_size,
                              hipStream_t stream) {
    (void)in_sizes; (void)n_in; (void)out_size; (void)ws_size;
    const int*   x    = (const int*)d_in[0];
    const int*   mask = (const int*)d_in[1];
    const float* emb  = (const float*)d_in[2];
    const float* Wih[4] = {(const float*)d_in[3], (const float*)d_in[6],
                           (const float*)d_in[9], (const float*)d_in[12]};
    const float* Whh[4] = {(const float*)d_in[4], (const float*)d_in[7],
                           (const float*)d_in[10], (const float*)d_in[13]};
    const float* bb[4]  = {(const float*)d_in[5], (const float*)d_in[8],
                           (const float*)d_in[11], (const float*)d_in[14]};
    const float* W1 = (const float*)d_in[15];
    const float* b1 = (const float*)d_in[16];
    const float* W2 = (const float*)d_in[17];
    const float* b2 = (const float*)d_in[18];

    char* ws = (char*)d_ws;
    size_t off = 0;
    int* lengths = (int*)(ws + off); off += 4096;
    unsigned short* WihP[4];
    for (int i = 0; i < 4; i++) { WihP[i] = (unsigned short*)(ws + off); off += (size_t)NP * KP * 2; }
    unsigned short* WhhP[4];
    for (int i = 0; i < 4; i++) { WhhP[i] = (unsigned short*)(ws + off); off += (size_t)NP * HKP * 2; }
    float* biasP[4];
    for (int i = 0; i < 4; i++) { biasP[i] = (float*)(ws + off); off += (size_t)NP * 4; }
    unsigned short* xe = (unsigned short*)(ws + off); off += (size_t)Bn * Sn * KP * 2;  // aliases h1
    float* Gf = (float*)(ws + off); off += (size_t)Bn * Sn * NP * 4;
    float* Gb = (float*)(ws + off); off += (size_t)Bn * Sn * NP * 4;
    float* lastb = (float*)(ws + off); off += (size_t)Bn * LASTP * 4;

    const int M = Bn * Sn;

    k_lengths<<<Bn, 256, 0, stream>>>(mask, lengths);

    for (int i = 0; i < 4; i++) {
        k_padw<<<(NP * KP + 255) / 256, 256, 0, stream>>>(Wih[i], WihP[i], 600, Dn, KP);
        k_padw<<<(NP * HKP + 255) / 256, 256, 0, stream>>>(Whh[i], WhhP[i], 600, Hn, HKP);
        k_padb<<<(NP + 255) / 256, 256, 0, stream>>>(bb[i], biasP[i]);
    }

    long embTotal = (long)M * KP;
    k_embed<<<(unsigned)((embTotal + 255) / 256), 256, 0, stream>>>(x, emb, xe, embTotal);

    dim3 ggrid(M / 128, NP / 128);
    // layer 0 input gates
    k_gemm_in<<<ggrid, 256, 0, stream>>>(xe, WihP[0], biasP[0], Gf, M);
    k_gemm_in<<<ggrid, 256, 0, stream>>>(xe, WihP[1], biasP[1], Gb, M);

    // zero h1 (aliases xe; safe after layer-0 GEMMs due to stream order)
    long zw = (long)M * KP * 2 / 4;
    k_zero32<<<(unsigned)((zw + 255) / 256), 256, 0, stream>>>((unsigned int*)xe, zw);

    (void)hipFuncSetAttribute(reinterpret_cast<const void*>(k_scan),
                              hipFuncAttributeMaxDynamicSharedMemorySize, SMEM_TOTAL);
    dim3 sgrid(Bn / 16, 2);
    k_scan<<<sgrid, 256, SMEM_TOTAL, stream>>>(Gf, Gb, WhhP[0], WhhP[1], lengths, xe, lastb, 0);

    // layer 1 input gates (input = h1, same padded K)
    k_gemm_in<<<ggrid, 256, 0, stream>>>(xe, WihP[2], biasP[2], Gf, M);
    k_gemm_in<<<ggrid, 256, 0, stream>>>(xe, WihP[3], biasP[3], Gb, M);

    k_scan<<<sgrid, 256, SMEM_TOTAL, stream>>>(Gf, Gb, WhhP[2], WhhP[3], lengths, xe, lastb, 1);

    k_head<<<Bn, 320, 0, stream>>>(lastb, W1, b1, W2, b2, (float*)d_out);
}
